// Res_up_62723702391726
// MI455X (gfx1250) — matrix operand order repaired and checked by reading, not compiled, not run on hardware
//
#include <hip/hip_runtime.h>
#include <math.h>

typedef __attribute__((ext_vector_type(16))) __bf16 v16bf;
typedef __attribute__((ext_vector_type(8)))  float  v8f;

#define NC   4096
#define NF   16384
#define CIN  128
#define COUT 128
#define CH   64
#define EC   65536
#define EF   262144
#define EPS_BN 1e-5f

__device__ __forceinline__ unsigned short f2bf(float f) {
  unsigned int u = __float_as_uint(f);
  u += 0x7fffu + ((u >> 16) & 1u);       // round-to-nearest-even
  return (unsigned short)(u >> 16);
}
// branch-free SELU: sc*(max(v,0) + al*(exp(min(v,0))-1))
__device__ __forceinline__ float selu_f(float v) {
  const float sc = 1.0507009873554805f, al = 1.6732632423543772f;
  return sc * (fmaxf(v, 0.f) + al * (__expf(fminf(v, 0.f)) - 1.f));
}

// ---------------------------------------------------------------- utilities
__global__ void zero_kernel(float* __restrict__ p, int n) {
  int t = blockIdx.x * blockDim.x + threadIdx.x;
  if (t < n) p[t] = 0.f;
}

// Pack f32 weight [K x Co] into bf16 WMMA B-fragment order:
// pk[((kt*(Co/16)+nt)*32 + lane)*16 + e]  holds  W[kt*32 + kk][nt*16 + n]
// with n=lane&15, hi=lane>>4, kk = (e&7) + hi*8 + ((e>>3)<<4).
__global__ void pack_w_kernel(const float* __restrict__ W,
                              unsigned short* __restrict__ pk, int K, int Co) {
  int t = blockIdx.x * blockDim.x + threadIdx.x;
  if (t >= K * Co) return;
  int e    = t & 15;
  int lane = (t >> 4) & 31;
  int tile = t >> 9;
  int ntile = Co >> 4;
  int nt = tile % ntile;
  int kt = tile / ntile;
  int n  = lane & 15, hi = lane >> 4;
  int kk = (e & 7) + hi * 8 + ((e >> 3) << 4);
  pk[t] = f2bf(W[(kt * 32 + kk) * Co + nt * 16 + n]);
}

// ---------------------------------------------------------------- kNN (k=3)
__global__ void knn_kernel(const float* __restrict__ pos_c,
                           const float* __restrict__ pos_f,
                           int* __restrict__ idx3, float* __restrict__ w4) {
  __shared__ float sp[2 * NC];                       // 32 KB of 320 KB WGP LDS
  for (int t = threadIdx.x; t < 2 * NC; t += blockDim.x) sp[t] = pos_c[t];
  __syncthreads();
  int n = blockIdx.x * blockDim.x + threadIdx.x;
  float px = pos_f[2 * n], py = pos_f[2 * n + 1];
  float b0 = 3.4e38f, b1 = 3.4e38f, b2 = 3.4e38f;
  int   i0 = 0, i1 = 0, i2 = 0;
  for (int j = 0; j < NC; ++j) {
    float dx = px - sp[2 * j], dy = py - sp[2 * j + 1];
    float d = dx * dx + dy * dy;
    if (d < b0)      { b2 = b1; i2 = i1; b1 = b0; i1 = i0; b0 = d; i0 = j; }
    else if (d < b1) { b2 = b1; i2 = i1; b1 = d;  i1 = j; }
    else if (d < b2) { b2 = d;  i2 = j; }
  }
  float w0 = 1.f / fmaxf(b0, 1e-16f);
  float w1 = 1.f / fmaxf(b1, 1e-16f);
  float w2 = 1.f / fmaxf(b2, 1e-16f);
  idx3[3 * n] = i0; idx3[3 * n + 1] = i1; idx3[3 * n + 2] = i2;
  w4[4 * n] = w0; w4[4 * n + 1] = w1; w4[4 * n + 2] = w2; w4[4 * n + 3] = w0 + w1 + w2;
}

__global__ void interp_kernel(const float* __restrict__ X, int C,
                              const int* __restrict__ idx3,
                              const float* __restrict__ w4,
                              float* __restrict__ out, int total) {
  int t = blockIdx.x * blockDim.x + threadIdx.x;
  if (t >= total) return;
  int n = t / C, c = t % C;
  int i0 = idx3[3 * n], i1 = idx3[3 * n + 1], i2 = idx3[3 * n + 2];
  float w0 = w4[4 * n], w1 = w4[4 * n + 1], w2 = w4[4 * n + 2], ws = w4[4 * n + 3];
  out[t] = (w0 * X[i0 * C + c] + w1 * X[i1 * C + c] + w2 * X[i2 * C + c]) / ws;
}

// --------------------------------------------------- fragment-order A staging
// Af[((rt*nkt + kt)*32 + lane)*16 + e] = A[rt*16 + m][kt*32 + kk]
// with lane = m + 16*hi, hi = (kk>>3)&1, e = (kk&7) + ((kk>>4)<<3).
__device__ __forceinline__ void stage_a(unsigned short* Af, int r, int k,
                                        int K, float v) {
  int nkt = K >> 5;
  int rt = r >> 4, m = r & 15;
  int kt = k >> 5, kk = k & 31;
  int hi = (kk >> 3) & 1;
  int e  = (kk & 7) + ((kk >> 4) << 3);
  Af[(((rt * nkt + kt) * 32) + (m + (hi << 4))) * 16 + e] = f2bf(v);
}

// 32x16xK WMMA pipeline: 2 row tiles share each B fragment (1 b128-pair load).
__device__ __forceinline__ void wmma_pair(const unsigned short* Af,
                                          const unsigned short* __restrict__ Wpk,
                                          int K, int Co, int nt, int lane,
                                          v8f& acc0, v8f& acc1) {
  const int nkt = K >> 5;
  for (int kt = 0; kt < nkt; ++kt) {
    v16bf b  = *(const v16bf*)(Wpk + (((kt * (Co >> 4)) + nt) * 32 + lane) * 16);
    v16bf a0 = *(const v16bf*)(Af + ((kt * 32) + lane) * 16);
    v16bf a1 = *(const v16bf*)(Af + (((nkt + kt) * 32) + lane) * 16);
    acc0 = __builtin_amdgcn_wmma_f32_16x16x32_bf16(false, a0, false, b,
                                                   (short)0, acc0, false, false);
    acc1 = __builtin_amdgcn_wmma_f32_16x16x32_bf16(false, a1, false, b,
                                                   (short)0, acc1, false, false);
  }
}

// edge MLP: m = selu([x[src]||x[dst]] @ We + be); agg[dst] += m  (segment_sum)
__global__ void edge_mlp_kernel(const float* __restrict__ X, int C,
                                const int* __restrict__ ei0,
                                const int* __restrict__ ei1,
                                const unsigned short* __restrict__ Wpk,
                                const float* __restrict__ bias, int Co,
                                float* __restrict__ agg) {
  __shared__ unsigned short Af[8192];   // 32 rows x up to K=256, fragment order
  __shared__ int ssrc[32];
  __shared__ int sdst[32];
  const int K = 2 * C;
  int base = blockIdx.x * 32;
  if (threadIdx.x < 32) {
    ssrc[threadIdx.x] = ei0[base + threadIdx.x];
    sdst[threadIdx.x] = ei1[base + threadIdx.x];
  }
  __syncthreads();
  for (int t = threadIdx.x; t < 32 * K; t += blockDim.x) {
    int r = t / K, k = t % K;
    int node = (k < C) ? ssrc[r] : sdst[r];
    int kk   = (k < C) ? k : (k - C);
    stage_a(Af, r, k, K, X[node * C + kk]);
  }
  __syncthreads();
  int lane = threadIdx.x & 31;
  int nt   = threadIdx.x >> 5;          // one 16-col tile per wave
  v8f acc0 = {}, acc1 = {};
  wmma_pair(Af, Wpk, K, Co, nt, lane, acc0, acc1);
  int m = lane & 15, hi = lane >> 4;
  int n = nt * 16 + m;
  float bv = bias[n];
#pragma unroll
  for (int i = 0; i < 8; ++i) {
    int r0 = i + hi * 8;
    atomicAdd(&agg[sdst[r0] * Co + n],      selu_f(acc0[i] + bv));
    atomicAdd(&agg[sdst[16 + r0] * Co + n], selu_f(acc1[i] + bv));
  }
}

// node MLP: out = selu([X0 || X1] @ Wn + bn)
__global__ void node_mlp_kernel(const float* __restrict__ X0, int C0,
                                const float* __restrict__ X1, int C1,
                                const unsigned short* __restrict__ Wpk,
                                const float* __restrict__ bias, int Co,
                                float* __restrict__ out) {
  __shared__ unsigned short Af[8192];
  const int K = C0 + C1;
  int base = blockIdx.x * 32;
  for (int t = threadIdx.x; t < 32 * K; t += blockDim.x) {
    int r = t / K, k = t % K;
    int node = base + r;
    float v = (k < C0) ? X0[node * C0 + k] : X1[node * C1 + (k - C0)];
    stage_a(Af, r, k, K, v);
  }
  __syncthreads();
  int lane = threadIdx.x & 31;
  int nt   = threadIdx.x >> 5;
  v8f acc0 = {}, acc1 = {};
  wmma_pair(Af, Wpk, K, Co, nt, lane, acc0, acc1);
  int m = lane & 15, hi = lane >> 4;
  int n = nt * 16 + m;
  float bv = bias[n];
#pragma unroll
  for (int i = 0; i < 8; ++i) {
    int r0 = i + hi * 8;
    out[(base + r0) * Co + n]      = selu_f(acc0[i] + bv);
    out[(base + 16 + r0) * Co + n] = selu_f(acc1[i] + bv);
  }
}

// ------------------------------------------------------- residual + BN + SELU
__global__ void bn_stats_kernel(const float* __restrict__ a,
                                const float* __restrict__ b,
                                float* __restrict__ stats) {
  int c   = threadIdx.x & 127;
  int sub = threadIdx.x >> 7;               // 0..1
  int r0  = blockIdx.x * 64 + sub;
  float s = 0.f, sq = 0.f;
  for (int r = r0; r < blockIdx.x * 64 + 64; r += 2) {
    float v = a[r * 128 + c] + b[r * 128 + c];
    s += v; sq += v * v;
  }
  atomicAdd(&stats[c], s);
  atomicAdd(&stats[128 + c], sq);
}

__global__ void bn_final_kernel(const float* __restrict__ a,
                                const float* __restrict__ b,
                                const float* __restrict__ stats,
                                const float* __restrict__ gamma,
                                const float* __restrict__ beta,
                                float* __restrict__ out) {
  int t = blockIdx.x * blockDim.x + threadIdx.x;
  int c = t & 127;
  float invN = 1.f / (float)NF;
  float mu  = stats[c] * invN;
  float var = stats[128 + c] * invN - mu * mu;
  float v = a[t] + b[t];
  out[t] = selu_f((v - mu) * rsqrtf(var + EPS_BN) * gamma[c] + beta[c]);
}

// --------------------------------------------------------------------- launch
extern "C" void kernel_launch(void* const* d_in, const int* in_sizes, int n_in,
                              void* d_out, int out_size, void* d_ws, size_t ws_size,
                              hipStream_t stream) {
  (void)in_sizes; (void)n_in; (void)out_size; (void)ws_size;
  const float* x     = (const float*)d_in[0];
  const float* pos_c = (const float*)d_in[1];
  const float* pos_f = (const float*)d_in[2];
  const int*   ei_c  = (const int*)d_in[3];
  const int*   ei_f  = (const int*)d_in[4];
  const float* We1 = (const float*)d_in[5];
  const float* be1 = (const float*)d_in[6];
  const float* Wn1 = (const float*)d_in[7];
  const float* bn1 = (const float*)d_in[8];
  const float* We2 = (const float*)d_in[9];
  const float* be2 = (const float*)d_in[10];
  const float* Wn2 = (const float*)d_in[11];
  const float* bn2 = (const float*)d_in[12];
  const float* WeS = (const float*)d_in[13];
  const float* beS = (const float*)d_in[14];
  const float* WnS = (const float*)d_in[15];
  const float* bnS = (const float*)d_in[16];
  const float* gamma = (const float*)d_in[17];
  const float* beta  = (const float*)d_in[18];

  size_t off = 0;
  char* base = (char*)d_ws;
  auto alloc = [&](size_t bytes) -> void* {
    void* p = base + off;
    off += (bytes + 255) & ~(size_t)255;
    return p;
  };
  unsigned short* wWe1 = (unsigned short*)alloc((size_t)256 * 64 * 2);
  unsigned short* wWn1 = (unsigned short*)alloc((size_t)192 * 64 * 2);
  unsigned short* wWe2 = (unsigned short*)alloc((size_t)128 * 128 * 2);
  unsigned short* wWn2 = (unsigned short*)alloc((size_t)192 * 128 * 2);
  unsigned short* wWeS = (unsigned short*)alloc((size_t)256 * 128 * 2);
  unsigned short* wWnS = (unsigned short*)alloc((size_t)256 * 128 * 2);
  int*   idx3  = (int*)  alloc((size_t)NF * 3 * 4);
  float* w4    = (float*)alloc((size_t)NF * 4 * 4);
  float* x_up  = (float*)alloc((size_t)NF * CIN * 4);
  float* h1    = (float*)alloc((size_t)NC * CH * 4);
  float* h_up  = (float*)alloc((size_t)NF * CH * 4);
  float* h2    = (float*)alloc((size_t)NF * COUT * 4);
  float* xskip = (float*)alloc((size_t)NF * COUT * 4);
  float* agg   = (float*)alloc((size_t)NF * COUT * 4);
  float* stats = (float*)alloc(256 * 4);

  // weights -> bf16, pre-packed in WMMA B-fragment order
  pack_w_kernel<<<(256*64  + 255)/256, 256, 0, stream>>>(We1, wWe1, 256, 64);
  pack_w_kernel<<<(192*64  + 255)/256, 256, 0, stream>>>(Wn1, wWn1, 192, 64);
  pack_w_kernel<<<(128*128 + 255)/256, 256, 0, stream>>>(We2, wWe2, 128, 128);
  pack_w_kernel<<<(192*128 + 255)/256, 256, 0, stream>>>(Wn2, wWn2, 192, 128);
  pack_w_kernel<<<(256*128 + 255)/256, 256, 0, stream>>>(WeS, wWeS, 256, 128);
  pack_w_kernel<<<(256*128 + 255)/256, 256, 0, stream>>>(WnS, wWnS, 256, 128);

  // kNN once (positions fixed), reused for both interpolations
  knn_kernel<<<NF / 256, 256, 0, stream>>>(pos_c, pos_f, idx3, w4);
  interp_kernel<<<(NF * CIN) / 256, 256, 0, stream>>>(x, CIN, idx3, w4, x_up, NF * CIN);

  // skip branch: mpl_skip on fine graph with x_up
  zero_kernel<<<(NF * COUT) / 256, 256, 0, stream>>>(agg, NF * COUT);
  edge_mlp_kernel<<<EF / 32, 2 * COUT, 0, stream>>>(x_up, CIN, ei_f, ei_f + EF,
                                                    wWeS, beS, COUT, agg);
  node_mlp_kernel<<<NF / 32, 2 * COUT, 0, stream>>>(x_up, CIN, agg, COUT,
                                                    wWnS, bnS, COUT, xskip);

  // main branch: mpl1 (coarse) -> upsample -> mpl2 (fine)
  zero_kernel<<<(NC * CH) / 256, 256, 0, stream>>>(agg, NC * CH);
  edge_mlp_kernel<<<EC / 32, 2 * CH, 0, stream>>>(x, CIN, ei_c, ei_c + EC,
                                                  wWe1, be1, CH, agg);
  node_mlp_kernel<<<NC / 32, 2 * CH, 0, stream>>>(x, CIN, agg, CH,
                                                  wWn1, bn1, CH, h1);
  interp_kernel<<<(NF * CH) / 256, 256, 0, stream>>>(h1, CH, idx3, w4, h_up, NF * CH);

  zero_kernel<<<(NF * COUT) / 256, 256, 0, stream>>>(agg, NF * COUT);
  edge_mlp_kernel<<<EF / 32, 2 * COUT, 0, stream>>>(h_up, CH, ei_f, ei_f + EF,
                                                    wWe2, be2, COUT, agg);
  node_mlp_kernel<<<NF / 32, 2 * COUT, 0, stream>>>(h_up, CH, agg, COUT,
                                                    wWn2, bn2, COUT, h2);

  // residual + BatchNorm(train) + SELU
  zero_kernel<<<1, 256, 0, stream>>>(stats, 256);
  bn_stats_kernel<<<NF / 64, 256, 0, stream>>>(h2, xskip, stats);
  bn_final_kernel<<<(NF * COUT) / 256, 256, 0, stream>>>(h2, xskip, stats,
                                                         gamma, beta, (float*)d_out);
}
